// PhysiologicalSignalDetector_23673859735564
// MI455X (gfx1250) — matrix operand order repaired and checked by reading, not compiled, not run on hardware
//
#include <hip/hip_runtime.h>
#include <hip/hip_bf16.h>
#include <math.h>

typedef __attribute__((ext_vector_type(16))) _Float16 v16h;
typedef __attribute__((ext_vector_type(8)))  _Float16 v8h;
typedef __attribute__((ext_vector_type(8)))  float    v8f;

#define PI_F 3.14159f

// ---------------------------------------------------------------------------
// Workspace layout (bytes), all offsets 256B-aligned
// ---------------------------------------------------------------------------
#define OFF_H1P   0ull                 // f16 [384][32][32][64]   50331648
#define OFF_H2P   50331648ull          // f16 [384][16][16][128]  25165824
#define OFF_BMAT1 75497472ull          // f16 [4][32][16]             4096
#define OFF_BMAT2 75501568ull          // f16 [8][18][32][16]       147456
#define OFF_SB1   75649024ull          // f32 scale[64]+bias[64]       512
#define OFF_SB2   75649536ull          // f32 scale[128]+bias[128]    1024
#define OFF_FEAT  75650560ull          // f32 [384][128]            196608
#define OFF_XT0   75847168ull          // f32 [96][4][128]          196608
#define OFF_XT1   76043776ull          // f32 [96][4][128]          196608
#define OFF_XT2   76240384ull          // f32 [96][4][128]          196608
#define OFF_GREEN 76436992ull          // f32 [384]                   1536
#define OFF_FENC  76438528ull          // f32 [4][64]                 1024
#define OFF_BMATL 76439552ull          // f16 [2][32][4][32][16]    262144
#define OFF_GBIAS 76701696ull          // f32 [2][512]                4096
#define OFF_GX    76705792ull          // f32 [96][4][512]          786432
                                       // end: 77492224 (~77.5 MB)

// Per-lane K index inside a 16-bit 16x32 WMMA A fragment (wave32):
//   lanes 0-15 : j -> K = j (j<8) else j+8     (K in {0..7,16..23})
//   lanes 16-31: j -> K = j+8 (j<8) else j+16  (K in {8..15,24..31})
__device__ __forceinline__ int a_kidx(int lhalf, int j) {
    return (lhalf == 0) ? ((j < 8) ? j : j + 8)
                        : ((j < 8) ? j + 8 : j + 16);
}

// Build a v16h A fragment from 32 contiguous f16 at `base` (two aligned
// 16B runs per lane): lane<16 reads halves {0..7,16..23}, lane>=16 {8..15,24..31}
__device__ __forceinline__ v16h a_frag_contig(const _Float16* base, int lhalf) {
    v8h lo = *(const v8h*)(base + (lhalf ? 8 : 0));
    v8h hi = *(const v8h*)(base + (lhalf ? 24 : 16));
    return __builtin_shufflevector(lo, hi, 0, 1, 2, 3, 4, 5, 6, 7,
                                   8, 9, 10, 11, 12, 13, 14, 15);
}

// ---------------------------------------------------------------------------
// Prep: swizzle conv + LSTM-input weights to WMMA B layout (f16), fold BN,
// fold LSTM biases.
// ---------------------------------------------------------------------------
__global__ void prep_kernel(const float* __restrict__ c1w, const float* __restrict__ c1b,
                            const float* __restrict__ g1, const float* __restrict__ be1,
                            const float* __restrict__ m1, const float* __restrict__ v1,
                            const float* __restrict__ c2w, const float* __restrict__ c2b,
                            const float* __restrict__ g2, const float* __restrict__ be2,
                            const float* __restrict__ m2, const float* __restrict__ v2,
                            const float* __restrict__ wih, const float* __restrict__ bih,
                            const float* __restrict__ bhh,
                            _Float16* __restrict__ bmat1, _Float16* __restrict__ bmat2,
                            float* __restrict__ sb1, float* __restrict__ sb2,
                            _Float16* __restrict__ bmatL, float* __restrict__ gbias) {
    const int NB2 = 8 * 18 * 32 * 16;     // 73728
    const int NB1 = 4 * 32 * 16;          // 2048
    const int NBL = 2 * 32 * 4 * 32 * 16; // 131072
    int idx = blockIdx.x * 256 + threadIdx.x;
    if (idx < NB2) {
        // bmat2[ntile][s][lane][j] : K order k = (ky*3+kx)*64 + c
        int j = idx & 15, lane = (idx >> 4) & 31;
        int s = (idx >> 9) % 18, ntile = idx / (16 * 32 * 18);
        int n = ntile * 16 + (lane & 15);
        int kl = (j < 8) ? 8 * (lane >> 4) + j : 8 + 8 * (lane >> 4) + j;      // B layout: lanes 0-15 K={0..7,16..23}
        int k = 32 * s + kl;                    // [0,576)
        int r = k >> 6, c = k & 63;
        bmat2[idx] = (_Float16)c2w[((size_t)n * 64 + c) * 9 + r];
    } else if (idx < NB2 + NB1) {
        // bmat1[ntile][lane][j] : k = (ky*3+kx)*3 + c, 27 valid pad->32
        int i = idx - NB2;
        int j = i & 15, lane = (i >> 4) & 31, ntile = i >> 9;
        int n = ntile * 16 + (lane & 15);
        int k = (j < 8) ? 8 * (lane >> 4) + j : 8 + 8 * (lane >> 4) + j;
        _Float16 v = (_Float16)0.f;
        if (k < 27) { int c = k % 3, r = k / 3; v = (_Float16)c1w[(size_t)n * 27 + c * 9 + r]; }
        bmat1[i] = v;
    } else if (idx < NB2 + NB1 + 64) {
        int ch = idx - NB2 - NB1;
        float inv = rsqrtf(v1[ch] + 1e-5f);
        float sc = g1[ch] * inv;
        sb1[ch] = sc;
        sb1[64 + ch] = be1[ch] + (c1b[ch] - m1[ch]) * sc;
    } else if (idx < NB2 + NB1 + 64 + 128) {
        int ch = idx - NB2 - NB1 - 64;
        float inv = rsqrtf(v2[ch] + 1e-5f);
        float sc = g2[ch] * inv;
        sb2[ch] = sc;
        sb2[128 + ch] = be2[ch] + (c2b[ch] - m2[ch]) * sc;
    } else if (idx < NB2 + NB1 + 64 + 128 + NBL) {
        // bmatL[layer][ntile(32)][s(4)][lane][j]; N = dir*256 + gate-row
        int i = idx - NB2 - NB1 - 64 - 128;
        int j = i & 15, lane = (i >> 4) & 31;
        int s = (i >> 9) & 3, nt = (i >> 11) & 31, L = (i >> 16) & 1;
        int n = nt * 16 + (lane & 15);          // 0..511
        int dir = n >> 8, row = n & 255;
        int kl = (j < 8) ? 8 * (lane >> 4) + j : 8 + 8 * (lane >> 4) + j;
        int k = 32 * s + kl;                    // [0,128)
        bmatL[i] = (_Float16)wih[(((size_t)(L * 2 + dir) * 256) + row) * 128 + k];
    } else if (idx < NB2 + NB1 + 64 + 128 + NBL + 2 * 512) {
        int i = idx - NB2 - NB1 - 64 - 128 - NBL;  // [0,1024)
        int L = i >> 9, n = i & 511;
        int dir = n >> 8, row = n & 255;
        gbias[i] = bih[(L * 2 + dir) * 256 + row] + bhh[(L * 2 + dir) * 256 + row];
    }
}

// ---------------------------------------------------------------------------
// Conv1 (3->64, 3x3 pad1 on 64x64) + BN + ReLU + maxpool2 via WMMA.
// Workgroup = 8 waves = 8 adjacent 4x4 blocks in one row, sharing a
// 6x34x3 f16 halo tile in LDS. Each wave builds one A fragment (K=27->32)
// and reuses it for all 4 channel tiles -> 4 WMMAs per A build.
// ---------------------------------------------------------------------------
__global__ void conv1_kernel(const float* __restrict__ x,
                             const _Float16* __restrict__ bmat1,
                             const float* __restrict__ sb1,
                             _Float16* __restrict__ h1p) {
    __shared__ alignas(16) _Float16 tile[6 * 34 * 3];   // [y][x][c]
    const int tid = threadIdx.x;
    const int bid = blockIdx.x;
    const int frame = bid >> 5;               // 32 groups per frame
    const int g = bid & 31;
    const int by = g >> 1, bxg = g & 1;       // row of blocks, half-row group

    __builtin_prefetch(bmat1, 0, 1);          // global_prefetch_b8 (weights)

    // cooperative halo stage (f32 -> f16, zero-padded borders)
    const float* xf = x + (size_t)frame * 3 * 4096;
    for (int i = tid; i < 6 * 34 * 3; i += 256) {
        int c = i % 3, rest = i / 3;
        int txx = rest % 34, tyy = rest / 34;
        int gy = 4 * by - 1 + tyy, gx = 32 * bxg - 1 + txx;
        float v = 0.f;
        if ((unsigned)gy < 64u && (unsigned)gx < 64u)
            v = xf[(size_t)c * 4096 + gy * 64 + gx];
        tile[i] = (_Float16)v;
    }
    __syncthreads();

    const int wave = tid >> 5, lane = tid & 31;
    const int lhalf = lane >> 4, l16 = lane & 15;
    // pixel m = window*4 + pos inside this wave's 4x4 block
    const int m = l16;
    const int w0 = m >> 2, p0 = m & 3;
    const int ly = 2 * (w0 >> 1) + (p0 >> 1);
    const int lx = 2 * (w0 & 1) + (p0 & 1);

    v16h a;
#pragma unroll
    for (int j = 0; j < 16; ++j) {
        int k = a_kidx(lhalf, j);
        _Float16 v = (_Float16)0.f;
        if (k < 27) {
            int c = k % 3, r = k / 3, ky = r / 3, kx = r % 3;
            v = tile[((ly + ky) * 34 + (4 * wave + lx + kx)) * 3 + c];
        }
        a[j] = v;
    }

    const int bxb = bxg * 8 + wave;           // this wave's block column
#pragma unroll
    for (int nt = 0; nt < 4; ++nt) {
        v16h bv = *(const v16h*)(bmat1 + ((size_t)(nt * 32 + lane)) * 16);
        v8f acc = {};
        acc = __builtin_amdgcn_wmma_f32_16x16x32_f16(false, a, false, bv,
                                                     (short)0, acc, false, false);
        const int ch = nt * 16 + l16;
        const float sc = sb1[ch], bi = sb1[64 + ch];
        float d[8];
#pragma unroll
        for (int r = 0; r < 8; ++r) { float t = acc[r] * sc + bi; d[r] = t > 0.f ? t : 0.f; }
#pragma unroll
        for (int wi = 0; wi < 2; ++wi) {
            int w = lhalf * 2 + wi;
            const float* dd = d + wi * 4;
            float mx = fmaxf(fmaxf(dd[0], dd[1]), fmaxf(dd[2], dd[3]));
            int py = 2 * by + (w >> 1), px = 2 * bxb + (w & 1);
            h1p[(((size_t)frame * 32 + py) * 32 + px) * 64 + ch] = (_Float16)mx;
        }
    }
}

// ---------------------------------------------------------------------------
// Conv2 (64->128, 3x3 pad1 on 32x32) + BN + ReLU + maxpool2 via WMMA.
// Workgroup = TWO adjacent 4x4 blocks x 8 channel tiles sharing a 6x10x64
// f16 halo tile in LDS. Each wave reuses every B fragment for two A
// fragments -> 36 WMMAs per wave, half the B traffic per WMMA.
// K = 576 = 18 steps of 32 (k = (ky*3+kx)*64 + c); A = two ds_load_b128.
// ---------------------------------------------------------------------------
__global__ void conv2_kernel(const _Float16* __restrict__ h1p,
                             const _Float16* __restrict__ bmat2,
                             const float* __restrict__ sb2,
                             _Float16* __restrict__ h2p) {
    __shared__ alignas(16) _Float16 tile[6 * 10 * 64];   // 7680 B
    const int tid = threadIdx.x;
    const int bid = blockIdx.x;
    const int frame = bid >> 5;               // 32 block-pairs per frame
    const int p = bid & 31;
    const int by = p >> 2, bxp = (p & 3) * 2; // blocks (by,bxp) and (by,bxp+1)
    const int y0 = 4 * by - 1, x0 = 4 * bxp - 1;

    const int wave = tid >> 5, lane = tid & 31;
    const int lhalf = lane >> 4, l16 = lane & 15;
    const int ntile = wave;
    const _Float16* bp = bmat2 + ((size_t)(ntile * 18) * 32 + lane) * 16;
    __builtin_prefetch(bp, 0, 1);             // global_prefetch_b8 (B stream)

    // cooperative halo stage for both blocks (zero-padded borders)
    const _Float16* hf = h1p + (size_t)frame * 32 * 32 * 64;
    for (int i = tid; i < 6 * 10 * 64; i += 256) {
        int c = i & 63, sp = i >> 6, ty = sp / 10, tx = sp % 10;
        int iy = y0 + ty, ix = x0 + tx;
        _Float16 v = (_Float16)0.f;
        if ((unsigned)iy < 32u && (unsigned)ix < 32u)
            v = hf[((size_t)iy * 32 + ix) * 64 + c];
        tile[i] = v;
    }
    __syncthreads();

    const int m = l16;
    const int w0 = m >> 2, p0 = m & 3;
    const int ly = 2 * (w0 >> 1) + (p0 >> 1);
    const int lx = 2 * (w0 & 1) + (p0 & 1);

    v8f acc0 = {}, acc1 = {};
#pragma unroll
    for (int s = 0; s < 18; ++s) {
        v16h bv = *(const v16h*)(bp + (size_t)s * 512);
        int r = s >> 1;                       // filter tap fixed per step
        int ky = r / 3, kx = r % 3;
        const _Float16* tp0 = &tile[(((ly + ky) * 10) + (lx + kx)) * 64 + (s & 1) * 32];
        v16h a0 = a_frag_contig(tp0, lhalf);
        v16h a1 = a_frag_contig(tp0 + 4 * 64, lhalf);   // second block: x+4
        acc0 = __builtin_amdgcn_wmma_f32_16x16x32_f16(false, a0, false, bv,
                                                      (short)0, acc0, false, false);
        acc1 = __builtin_amdgcn_wmma_f32_16x16x32_f16(false, a1, false, bv,
                                                      (short)0, acc1, false, false);
    }

    const int ch = ntile * 16 + l16;
    const float sc = sb2[ch], bi = sb2[128 + ch];
#pragma unroll
    for (int q = 0; q < 2; ++q) {
        const v8f& acc = q ? acc1 : acc0;
        float d[8];
#pragma unroll
        for (int r = 0; r < 8; ++r) { float t = acc[r] * sc + bi; d[r] = t > 0.f ? t : 0.f; }
#pragma unroll
        for (int wi = 0; wi < 2; ++wi) {
            int w = lhalf * 2 + wi;
            const float* dd = d + wi * 4;
            float mx = fmaxf(fmaxf(dd[0], dd[1]), fmaxf(dd[2], dd[3]));
            int py = 2 * by + (w >> 1), px = 2 * (bxp + q) + (w & 1);
            h2p[(((size_t)frame * 16 + py) * 16 + px) * 128 + ch] = (_Float16)mx;
        }
    }
}

// ---------------------------------------------------------------------------
// Global average pool: feat[f][ch] = mean over 16x16 of h2p
// ---------------------------------------------------------------------------
__global__ void avgpool_kernel(const _Float16* __restrict__ h2p, float* __restrict__ feat) {
    int f = blockIdx.x, ch = threadIdx.x;     // 128 threads
    const _Float16* p = h2p + (size_t)f * 256 * 128 + ch;
    float s = 0.f;
    for (int i = 0; i < 256; ++i) s += (float)p[(size_t)i * 128];
    feat[(size_t)f * 128 + ch] = s * (1.f / 256.f);
}

// ---------------------------------------------------------------------------
// Temporal attention: weighted[t][b][c] = feat[f][c] * sigmoid(feat.attn_w+b)
// ---------------------------------------------------------------------------
__global__ void attn_kernel(const float* __restrict__ feat,
                            const float* __restrict__ attn_w,
                            const float* __restrict__ attn_b,
                            float* __restrict__ weighted) {
    int f = blockIdx.x, c = threadIdx.x;      // 128 threads
    __shared__ float red[128];
    float v = feat[(size_t)f * 128 + c];
    red[c] = v * attn_w[c];
    __syncthreads();
    for (int s = 64; s > 0; s >>= 1) { if (c < s) red[c] += red[c + s]; __syncthreads(); }
    float a = 1.f / (1.f + __expf(-(red[0] + attn_b[0])));
    int b = f / 96, t = f % 96;
    weighted[((size_t)t * 4 + b) * 128 + c] = v * a;
}

// ---------------------------------------------------------------------------
// LSTM input-gate GEMM via WMMA: gx[m][n] = X[m][:128] @ Wih^T + (bih+bhh)
// m = t*4+b (384 rows), n = dir*256 + gate-row (512 cols), K = 128 (4 steps).
// Workgroup = 1 m-tile x 8 n-tiles; 16x128 f16 X tile staged in LDS.
// ---------------------------------------------------------------------------
__global__ void gemm_gates_kernel(const float* __restrict__ xin,
                                  const _Float16* __restrict__ bmatL_l,
                                  const float* __restrict__ gbias_l,
                                  float* __restrict__ gx) {
    __shared__ alignas(16) _Float16 xtile[16 * 128];
    const int tid = threadIdx.x;
    const int bid = blockIdx.x;               // 24 m-tiles x 4 n-groups
    const int mtile = bid >> 2, ngrp = bid & 3;

    for (int i = tid; i < 16 * 128; i += 256) {
        int row = i >> 7, k = i & 127;
        xtile[i] = (_Float16)xin[((size_t)mtile * 16 + row) * 128 + k];
    }
    __syncthreads();

    const int wave = tid >> 5, lane = tid & 31;
    const int lhalf = lane >> 4, l16 = lane & 15;
    const int nt = ngrp * 8 + wave;           // 0..31

    v8f acc = {};
    const _Float16* bp = bmatL_l + ((size_t)(nt * 4) * 32 + lane) * 16;
#pragma unroll
    for (int s = 0; s < 4; ++s) {
        v16h bv = *(const v16h*)(bp + (size_t)s * 512);
        v16h a = a_frag_contig(&xtile[l16 * 128 + s * 32], lhalf);
        acc = __builtin_amdgcn_wmma_f32_16x16x32_f16(false, a, false, bv,
                                                     (short)0, acc, false, false);
    }

    const int n = nt * 16 + l16;
    const float bi = gbias_l[n];
#pragma unroll
    for (int r = 0; r < 8; ++r) {
        int mrow = mtile * 16 + r + lhalf * 8;
        gx[(size_t)mrow * 512 + n] = acc[r] + bi;
    }
}

// ---------------------------------------------------------------------------
// LSTM recurrence (torch gate order i,f,g,o). grid = 2 (dir),
// block = (64 hidden, 4 batch). W_hh staged once in LDS as f16 (32 KB),
// h in LDS, c in registers. Input gates come precomputed from gx.
// ---------------------------------------------------------------------------
__global__ void lstm_seq_kernel(const float* __restrict__ gx,
                                const float* __restrict__ whh,
                                int layer, float* __restrict__ xout) {
    __shared__ _Float16 wsh[256 * 64];        // 32 KB
    __shared__ float hsh[4][64];
    const int dir = blockIdx.x;
    const int j = threadIdx.x, b = threadIdx.y;
    const int tid = b * 64 + j;
    const float* W = whh + (size_t)(layer * 2 + dir) * 256 * 64;
    for (int i = tid; i < 256 * 64; i += 256) wsh[i] = (_Float16)W[i];
    hsh[b][j] = 0.f;
    __syncthreads();

    float cst = 0.f;
    for (int step = 0; step < 96; ++step) {
        int t = dir ? (95 - step) : step;
        const float* gxr = gx + ((size_t)t * 4 + b) * 512 + dir * 256;
        float g[4];
#pragma unroll
        for (int gi = 0; gi < 4; ++gi) {
            int row = gi * 64 + j;
            const _Float16* wr = wsh + row * 64;
            float s = gxr[row];
            for (int k = 0; k < 64; ++k) s += (float)wr[k] * hsh[b][k];
            g[gi] = s;
        }
        __syncthreads();                       // done reading hsh
        float ig = 1.f / (1.f + __expf(-g[0]));
        float fg = 1.f / (1.f + __expf(-g[1]));
        float gg = tanhf(g[2]);
        float og = 1.f / (1.f + __expf(-g[3]));
        cst = fg * cst + ig * gg;
        float hj = og * tanhf(cst);
        hsh[b][j] = hj;
        xout[((size_t)t * 4 + b) * 128 + dir * 64 + j] = hj;
        __syncthreads();                       // hsh updated before next step
    }
}

// ---------------------------------------------------------------------------
// Per-frame green-channel mean
// ---------------------------------------------------------------------------
__global__ void green_kernel(const float* __restrict__ x, float* __restrict__ green) {
    int f = blockIdx.x, tid = threadIdx.x;    // 256 threads
    const float* p = x + ((size_t)f * 3 + 1) * 4096;
    float s = 0.f;
    for (int i = tid; i < 4096; i += 256) s += p[i];
    __shared__ float red[256];
    red[tid] = s;
    __syncthreads();
    for (int st = 128; st > 0; st >>= 1) { if (tid < st) red[tid] += red[tid + st]; __syncthreads(); }
    if (tid == 0) green[f] = red[0] * (1.f / 4096.f);
}

// ---------------------------------------------------------------------------
// Frequency correlation + 2-layer frequency encoder. 1 block x 96 threads.
// ---------------------------------------------------------------------------
__global__ void freq_kernel(const float* __restrict__ green,
                            const float* __restrict__ fe_w1, const float* __restrict__ fe_b1,
                            const float* __restrict__ fe_w2, const float* __restrict__ fe_b2,
                            float* __restrict__ freq_enc) {
    int tid = threadIdx.x;                    // 96 threads
    __shared__ float det[96];
    __shared__ float corr[30];
    __shared__ float h1[64];
    __shared__ float red;
    for (int b = 0; b < 4; ++b) {
        const float* g = green + b * 96;
        if (tid == 0) { float s = 0.f; for (int t = 0; t < 96; ++t) s += g[t]; red = s / 96.f; }
        __syncthreads();
        det[tid] = g[tid] - red;
        __syncthreads();
        if (tid < 30) {
            float fr = (tid + 1) * 0.1f;
            float s = 0.f;
            for (int t = 0; t < 96; ++t) s += det[t] * __sinf((float)t * fr * 2.f * PI_F / 96.f);
            corr[tid] = fabsf(s);
        }
        __syncthreads();
        if (tid == 0) { float m = 0.f; for (int i = 0; i < 30; ++i) m = fmaxf(m, corr[i]); red = m; }
        __syncthreads();
        float mx = red;
        if (tid < 64) {
            float s = fe_b1[tid];
            for (int i = 0; i < 30; ++i) {
                float ff = (mx > 0.f) ? corr[i] / mx : corr[i];
                s += fe_w1[tid * 30 + i] * ff;
            }
            h1[tid] = s > 0.f ? s : 0.f;
        }
        __syncthreads();
        if (tid < 64) {
            float s = fe_b2[tid];
            for (int i = 0; i < 64; ++i) s += fe_w2[tid * 64 + i] * h1[i];
            freq_enc[b * 64 + tid] = s;
        }
        __syncthreads();
    }
}

// ---------------------------------------------------------------------------
// Heads: hrv/skin/resp MLPs -> combined[13] -> fc -> emb[4][128]
// ---------------------------------------------------------------------------
__global__ void head_kernel(const float* __restrict__ xt2, const float* __restrict__ freq_enc,
                            const float* __restrict__ hw1, const float* __restrict__ hb1,
                            const float* __restrict__ hw2, const float* __restrict__ hb2,
                            const float* __restrict__ hw3, const float* __restrict__ hb3,
                            const float* __restrict__ sw1, const float* __restrict__ sb1_,
                            const float* __restrict__ sw2, const float* __restrict__ sb2_,
                            const float* __restrict__ sw3, const float* __restrict__ sb3_,
                            const float* __restrict__ rw1, const float* __restrict__ rb1,
                            const float* __restrict__ rw2, const float* __restrict__ rb2,
                            const float* __restrict__ rw3, const float* __restrict__ rb3,
                            const float* __restrict__ fw1, const float* __restrict__ fb1,
                            const float* __restrict__ fw2, const float* __restrict__ fb2,
                            float* __restrict__ out) {
    int tid = threadIdx.x;                    // 128 threads
    __shared__ float physio[192];
    __shared__ float h64[64];
    __shared__ float h32[32];
    __shared__ float comb[13];
    __shared__ float fch[128];
    for (int b = 0; b < 4; ++b) {
        physio[tid] = xt2[((size_t)95 * 4 + b) * 128 + tid];
        if (tid < 64) physio[128 + tid] = freq_enc[b * 64 + tid];
        __syncthreads();
        if (tid < 64) { float s = hb1[tid]; for (int i = 0; i < 192; ++i) s += hw1[tid * 192 + i] * physio[i]; h64[tid] = s > 0.f ? s : 0.f; }
        __syncthreads();
        if (tid < 32) { float s = hb2[tid]; for (int i = 0; i < 64; ++i) s += hw2[tid * 64 + i] * h64[i]; h32[tid] = s > 0.f ? s : 0.f; }
        __syncthreads();
        if (tid < 5)  { float s = hb3[tid]; for (int i = 0; i < 32; ++i) s += hw3[tid * 32 + i] * h32[i]; comb[tid] = s; }
        __syncthreads();
        if (tid < 64) { float s = sb1_[tid]; for (int i = 0; i < 128; ++i) s += sw1[tid * 128 + i] * physio[i]; h64[tid] = s > 0.f ? s : 0.f; }
        __syncthreads();
        if (tid < 32) { float s = sb2_[tid]; for (int i = 0; i < 64; ++i) s += sw2[tid * 64 + i] * h64[i]; h32[tid] = s > 0.f ? s : 0.f; }
        __syncthreads();
        if (tid < 4)  { float s = sb3_[tid]; for (int i = 0; i < 32; ++i) s += sw3[tid * 32 + i] * h32[i]; comb[5 + tid] = s; }
        __syncthreads();
        if (tid < 64) { float s = rb1[tid]; for (int i = 0; i < 128; ++i) s += rw1[tid * 128 + i] * physio[i]; h64[tid] = s > 0.f ? s : 0.f; }
        __syncthreads();
        if (tid < 32) { float s = rb2[tid]; for (int i = 0; i < 64; ++i) s += rw2[tid * 64 + i] * h64[i]; h32[tid] = s > 0.f ? s : 0.f; }
        __syncthreads();
        if (tid < 4)  { float s = rb3[tid]; for (int i = 0; i < 32; ++i) s += rw3[tid * 32 + i] * h32[i]; comb[9 + tid] = s; }
        __syncthreads();
        { float s = fb1[tid]; for (int i = 0; i < 13; ++i) s += fw1[tid * 13 + i] * comb[i]; fch[tid] = s > 0.f ? s : 0.f; }
        __syncthreads();
        { float s = fb2[tid]; for (int i = 0; i < 128; ++i) s += fw2[tid * 128 + i] * fch[i]; out[b * 128 + tid] = s; }
        __syncthreads();
    }
}

// ---------------------------------------------------------------------------
// Launch
// ---------------------------------------------------------------------------
extern "C" void kernel_launch(void* const* d_in, const int* in_sizes, int n_in,
                              void* d_out, int out_size, void* d_ws, size_t ws_size,
                              hipStream_t stream) {
    const float* x        = (const float*)d_in[0];
    const float* conv1_w  = (const float*)d_in[1];
    const float* conv1_b  = (const float*)d_in[2];
    const float* bn1_g    = (const float*)d_in[3];
    const float* bn1_b    = (const float*)d_in[4];
    const float* bn1_m    = (const float*)d_in[5];
    const float* bn1_v    = (const float*)d_in[6];
    const float* conv2_w  = (const float*)d_in[7];
    const float* conv2_b  = (const float*)d_in[8];
    const float* bn2_g    = (const float*)d_in[9];
    const float* bn2_b    = (const float*)d_in[10];
    const float* bn2_m    = (const float*)d_in[11];
    const float* bn2_v    = (const float*)d_in[12];
    const float* attn_w   = (const float*)d_in[13];
    const float* attn_b   = (const float*)d_in[14];
    const float* lstm_wih = (const float*)d_in[15];
    const float* lstm_whh = (const float*)d_in[16];
    const float* lstm_bih = (const float*)d_in[17];
    const float* lstm_bhh = (const float*)d_in[18];
    const float* fe_w1    = (const float*)d_in[19];
    const float* fe_b1    = (const float*)d_in[20];
    const float* fe_w2    = (const float*)d_in[21];
    const float* fe_b2    = (const float*)d_in[22];
    const float* hrv_w1   = (const float*)d_in[23];
    const float* hrv_b1   = (const float*)d_in[24];
    const float* hrv_w2   = (const float*)d_in[25];
    const float* hrv_b2   = (const float*)d_in[26];
    const float* hrv_w3   = (const float*)d_in[27];
    const float* hrv_b3   = (const float*)d_in[28];
    const float* skin_w1  = (const float*)d_in[29];
    const float* skin_b1  = (const float*)d_in[30];
    const float* skin_w2  = (const float*)d_in[31];
    const float* skin_b2  = (const float*)d_in[32];
    const float* skin_w3  = (const float*)d_in[33];
    const float* skin_b3  = (const float*)d_in[34];
    const float* resp_w1  = (const float*)d_in[35];
    const float* resp_b1  = (const float*)d_in[36];
    const float* resp_w2  = (const float*)d_in[37];
    const float* resp_b2  = (const float*)d_in[38];
    const float* resp_w3  = (const float*)d_in[39];
    const float* resp_b3  = (const float*)d_in[40];
    const float* fc_w1    = (const float*)d_in[41];
    const float* fc_b1    = (const float*)d_in[42];
    const float* fc_w2    = (const float*)d_in[43];
    const float* fc_b2    = (const float*)d_in[44];

    char* ws = (char*)d_ws;
    _Float16* h1p   = (_Float16*)(ws + OFF_H1P);
    _Float16* h2p   = (_Float16*)(ws + OFF_H2P);
    _Float16* bmat1 = (_Float16*)(ws + OFF_BMAT1);
    _Float16* bmat2 = (_Float16*)(ws + OFF_BMAT2);
    _Float16* bmatL = (_Float16*)(ws + OFF_BMATL);
    float* sb1   = (float*)(ws + OFF_SB1);
    float* sb2   = (float*)(ws + OFF_SB2);
    float* feat  = (float*)(ws + OFF_FEAT);
    float* xt0   = (float*)(ws + OFF_XT0);
    float* xt1   = (float*)(ws + OFF_XT1);
    float* xt2   = (float*)(ws + OFF_XT2);
    float* green = (float*)(ws + OFF_GREEN);
    float* fenc  = (float*)(ws + OFF_FENC);
    float* gbias = (float*)(ws + OFF_GBIAS);
    float* gx    = (float*)(ws + OFF_GX);
    float* out   = (float*)d_out;

    // 1) weight swizzle + BN/bias folds
    {
        int total = 8 * 18 * 32 * 16 + 4 * 32 * 16 + 64 + 128 + 2 * 32 * 4 * 32 * 16 + 2 * 512;
        prep_kernel<<<(total + 255) / 256, 256, 0, stream>>>(
            conv1_w, conv1_b, bn1_g, bn1_b, bn1_m, bn1_v,
            conv2_w, conv2_b, bn2_g, bn2_b, bn2_m, bn2_v,
            lstm_wih, lstm_bih, lstm_bhh,
            bmat1, bmat2, sb1, sb2, bmatL, gbias);
    }
    // 2) conv1+bn+relu+pool (WMMA, LDS-staged halo, 4 WMMAs per A build)
    conv1_kernel<<<384 * 32, 256, 0, stream>>>(x, bmat1, sb1, h1p);
    // 3) conv2+bn+relu+pool (WMMA, 2 tiles/wave, ds_load_b128 A fragments)
    conv2_kernel<<<384 * 32, 256, 0, stream>>>(h1p, bmat2, sb2, h2p);
    // 4) global average pool -> feat [384,128]
    avgpool_kernel<<<384, 128, 0, stream>>>(h2p, feat);
    // 5) attention -> weighted [96,4,128]
    attn_kernel<<<384, 128, 0, stream>>>(feat, attn_w, attn_b, xt0);
    // 6) green means (independent of 2-5)
    green_kernel<<<384, 256, 0, stream>>>(x, green);
    // 7) BiLSTM: per layer, WMMA input-gate GEMM then light recurrence
    gemm_gates_kernel<<<96, 256, 0, stream>>>(xt0, bmatL, gbias, gx);
    lstm_seq_kernel<<<2, dim3(64, 4), 0, stream>>>(gx, lstm_whh, 0, xt1);
    gemm_gates_kernel<<<96, 256, 0, stream>>>(xt1, bmatL + 65536, gbias + 512, gx);
    lstm_seq_kernel<<<2, dim3(64, 4), 0, stream>>>(gx, lstm_whh, 1, xt2);
    // 8) frequency encoder
    freq_kernel<<<1, 96, 0, stream>>>(green, fe_w1, fe_b1, fe_w2, fe_b2, fenc);
    // 9) heads -> emb [4,128]
    head_kernel<<<1, 128, 0, stream>>>(
        xt2, fenc,
        hrv_w1, hrv_b1, hrv_w2, hrv_b2, hrv_w3, hrv_b3,
        skin_w1, skin_b1, skin_w2, skin_b2, skin_w3, skin_b3,
        resp_w1, resp_b1, resp_w2, resp_b2, resp_w3, resp_b3,
        fc_w1, fc_b1, fc_w2, fc_b2, out);
}
